// ForwardSim_61383672594567
// MI455X (gfx1250) — matrix operand order repaired and checked by reading, not compiled, not run on hardware
//
#include <hip/hip_runtime.h>
#include <hip/hip_bf16.h>

#define B_TOT   32768
#define T_STEPS 30
#define DT_C    0.1f
#define WAVES   8

typedef _Float16 h16;
typedef __attribute__((ext_vector_type(16))) _Float16 v16h;
typedef __attribute__((ext_vector_type(8)))  _Float16 v8h;
typedef __attribute__((ext_vector_type(8)))  float    v8f;

#define W1_STRIDE 40    // 32 K rows padded (halves); 80B = 16B-aligned, banks spread
#define W2_STRIDE 136   // 128 K rows padded; 272B stride -> conflict-free B-frag loads
#define WH_STRIDE 136
#define X_STRIDE  40
#define H_STRIDE  136

union Frag16 { v16h v; v8h p[2]; };

// A-fragment style: halves[0..7] = K off_lo.., halves[8..15] = K off_hi..
__device__ __forceinline__ v16h ld_frag_split(const h16* base, int off_lo, int off_hi) {
  Frag16 f;
  f.p[0] = *(const v8h*)(base + off_lo);
  f.p[1] = *(const v8h*)(base + off_hi);
  return f.v;
}
// B-fragment style: 16 contiguous halves
__device__ __forceinline__ v16h ld_frag_contig(const h16* base, int off) {
  Frag16 f;
  f.p[0] = *(const v8h*)(base + off);
  f.p[1] = *(const v8h*)(base + off + 8);
  return f.v;
}

__global__ __launch_bounds__(256) void ForwardSim_kernel(
    const float* __restrict__ sampled_z, const float* __restrict__ idm_s,
    const float* __restrict__ merger_cs, const float* __restrict__ noise,
    const float* __restrict__ scaler_mean, const float* __restrict__ scaler_std,
    const float* __restrict__ W1g, const float* __restrict__ b1g,
    const float* __restrict__ W2g, const float* __restrict__ b2g,
    const float* __restrict__ Wmg, const float* __restrict__ bmg,
    const float* __restrict__ Wvg, const float* __restrict__ bvg,
    float* __restrict__ out)
{
  __shared__ __align__(16) h16 s_w1[128 * W1_STRIDE];   // W1 column-major, K padded to 32 (zeros)
  __shared__ __align__(16) h16 s_w2[128 * W2_STRIDE];   // W2 column-major
  __shared__ __align__(16) h16 s_wh[16  * WH_STRIDE];   // heads: col0=Wm col1=Wv, cols 2..15 zero
  __shared__ __align__(16) h16 s_x[WAVES][16 * X_STRIDE]; // per-wave x tile (16 x 32, f16)
  __shared__ __align__(16) h16 s_h[WAVES][16 * H_STRIDE]; // per-wave h tile (16 x 128, f16)

  const int tid = threadIdx.x;

  // ---- cooperative weight staging: f32 global -> f16 LDS (column-major, padded) ----
  for (int i = tid; i < 128 * W1_STRIDE; i += 256) s_w1[i] = (h16)0.f;
  for (int i = tid; i < 16  * WH_STRIDE; i += 256) s_wh[i] = (h16)0.f;
  __syncthreads();
  for (int i = tid; i < 11 * 128; i += 256) {           // W1 is [11][128] row-major
    int k = i >> 7, n = i & 127;
    s_w1[n * W1_STRIDE + k] = (h16)W1g[i];
  }
  for (int i = tid; i < 128 * 128; i += 256) {          // W2 is [128][128] row-major
    int k = i >> 7, n = i & 127;
    s_w2[n * W2_STRIDE + k] = (h16)W2g[i];
  }
  for (int i = tid; i < 128; i += 256) {
    s_wh[0 * WH_STRIDE + i] = (h16)Wmg[i];
    s_wh[1 * WH_STRIDE + i] = (h16)Wvg[i];
  }
  __syncthreads();

  const int wave = tid >> 5;
  const int lane = tid & 31;
  const int l    = lane & 15;
  const bool hiH = (lane >= 16);
  const int klo     = hiH ? 16 : 0;    // A-fragment low-K base
  const int kbB     = hiH ? 16 : 0;    // B-fragment K base
  const int rowbase = hiH ? 8 : 0;     // C/D fragment row base
  const int tile = blockIdx.x * WAVES + wave;
  const int b    = tile * 16 + l;      // batch row (state owned by lanes 0..15)

  h16* xs = s_x[wave];
  h16* hs = s_h[wave];

  // biases / scalers in registers (uniform-ish broadcast loads)
  float bias1[8], bias2[8];
#pragma unroll
  for (int n = 0; n < 8; ++n) { bias1[n] = b1g[n * 16 + l]; bias2[n] = b2g[n * 16 + l]; }
  const float bm_s = bmg[0], bv_s = bvg[0];
  float sm[6], si[6];
#pragma unroll
  for (int i = 0; i < 6; ++i) { sm[i] = scaler_mean[i]; si[i] = 1.f / scaler_std[i]; }

  // x tile static part: z (K=0..2) constant over t, K=11..31 zero-padded
  if (!hiH) {
#pragma unroll
    for (int k = 11; k < 32; ++k) xs[l * X_STRIDE + k] = (h16)0.f;
    const float* zp = sampled_z + (size_t)b * 3;
    xs[l * X_STRIDE + 0] = (h16)zp[0];
    xs[l * X_STRIDE + 1] = (h16)zp[1];
    xs[l * X_STRIDE + 2] = (h16)zp[2];
  }

  // carry init: ego_v = idm[b,0,0], ego_x = idm[b,0,3]
  float ego_v = 0.f, ego_x = 0.f;
  if (!hiH) {
    const float* r0 = idm_s + (size_t)b * T_STEPS * 11;
    ego_v = r0[0];
    ego_x = r0[3];
  }

  for (int t = 0; t < T_STEPS; ++t) {
    // ---- env features -> x tile (lanes 0..15, one batch row each) ----
    float nz = 0.f;
    if (!hiH) {
      const float* row = idm_s + ((size_t)b * T_STEPS + t) * 11;
      float f_v = row[1], m_v = row[2], f_x = row[4], m_x = row[5], m_e = row[10];
      float env0 = ego_v;
      float env1 = f_v;
      float env2 = ego_v - f_v;                                   // ef_dv
      float env3 = f_x - ego_x;                                   // ef_dx
      float env4 = (ego_v - m_v) * m_e;                           // em_dv (dummy 0)
      float env5 = (m_x - ego_x) * m_e + (1.f - m_e) * 100.f;     // em_dx (dummy 100)
      xs[l * X_STRIDE + 3] = (h16)((env0 - sm[0]) * si[0]);
      xs[l * X_STRIDE + 4] = (h16)((env1 - sm[1]) * si[1]);
      xs[l * X_STRIDE + 5] = (h16)((env2 - sm[2]) * si[2]);
      xs[l * X_STRIDE + 6] = (h16)((env3 - sm[3]) * si[3]);
      xs[l * X_STRIDE + 7] = (h16)((env4 - sm[4]) * si[4]);
      xs[l * X_STRIDE + 8] = (h16)((env5 - sm[5]) * si[5]);
      const float* mcp = merger_cs + ((size_t)b * T_STEPS + t) * 2;
      xs[l * X_STRIDE + 9]  = (h16)mcp[0];
      xs[l * X_STRIDE + 10] = (h16)mcp[1];
      nz = noise[(size_t)b * T_STEPS + t];
    }
    asm volatile("s_wait_dscnt 0x0" ::: "memory");  // intra-wave LDS handoff

    // ---- layer 1: [16x32] x [32x128] via 8 WMMAs ----
    v16h a1 = ld_frag_split(xs + l * X_STRIDE, klo, klo + 8);
    v8f acc[8];
#pragma unroll
    for (int n = 0; n < 8; ++n) {
      v16h bf = ld_frag_contig(s_w1 + (n * 16 + l) * W1_STRIDE, kbB);
      v8f cz = {};
      acc[n] = __builtin_amdgcn_wmma_f32_16x16x32_f16(false, a1, false, bf,
                                                      (short)0, cz, false, false);
    }
    // bias + relu -> h tile (f16) in LDS
#pragma unroll
    for (int n = 0; n < 8; ++n) {
#pragma unroll
      for (int r = 0; r < 8; ++r) {
        float v = acc[n][r] + bias1[n];
        v = v > 0.f ? v : 0.f;
        hs[(rowbase + r) * H_STRIDE + n * 16 + l] = (h16)v;
      }
    }
    asm volatile("s_wait_dscnt 0x0" ::: "memory");

    // ---- layer 2: [16x128] x [128x128] via 32 WMMAs (A-frags cached in regs) ----
    v16h afr[4];
#pragma unroll
    for (int kt = 0; kt < 4; ++kt)
      afr[kt] = ld_frag_split(hs + l * H_STRIDE, kt * 32 + klo, kt * 32 + klo + 8);
#pragma unroll
    for (int n = 0; n < 8; ++n) {
      v8f c = {};
#pragma unroll
      for (int kt = 0; kt < 4; ++kt) {
        v16h bf = ld_frag_contig(s_w2 + (n * 16 + l) * W2_STRIDE, kt * 32 + kbB);
        c = __builtin_amdgcn_wmma_f32_16x16x32_f16(false, afr[kt], false, bf,
                                                   (short)0, c, false, false);
      }
      acc[n] = c;
    }
#pragma unroll
    for (int n = 0; n < 8; ++n) {
#pragma unroll
      for (int r = 0; r < 8; ++r) {
        float v = acc[n][r] + bias2[n];
        v = v > 0.f ? v : 0.f;
        hs[(rowbase + r) * H_STRIDE + n * 16 + l] = (h16)v;
      }
    }
    asm volatile("s_wait_dscnt 0x0" ::: "memory");

    // ---- heads: [16x128] x [128x16] (cols 0,1 live) via 4 WMMAs ----
    v8f ch = {};
#pragma unroll
    for (int kt = 0; kt < 4; ++kt) {
      v16h af = ld_frag_split(hs + l * H_STRIDE, kt * 32 + klo, kt * 32 + klo + 8);
      v16h bf = ld_frag_contig(s_wh + l * WH_STRIDE, kt * 32 + kbB);
      ch = __builtin_amdgcn_wmma_f32_16x16x32_f16(false, af, false, bf,
                                                  (short)0, ch, false, false);
    }
    // exchange mean/logvar columns to row-owning lanes via LDS (reuse hs)
    float* ex = (float*)hs;
    if (l < 2) {
#pragma unroll
      for (int r = 0; r < 8; ++r) ex[l * 16 + rowbase + r] = ch[r];
    }
    asm volatile("s_wait_dscnt 0x0" ::: "memory");

    if (!hiH) {
      float mean = ex[l] + bm_s;
      float vlog = ex[16 + l] + bv_s;
      float var  = __expf(vlog);
      float act  = mean + var * nz;
      float v_next = ego_v + act * DT_C;
      float x_next = ego_x + v_next * DT_C + 0.5f * act * DT_C * DT_C;
      ego_v = v_next;
      ego_x = x_next;
      size_t o = (size_t)b * T_STEPS + t;
      const size_t BT = (size_t)B_TOT * T_STEPS;
      out[o]          = mean;
      out[BT + o]     = var;
      out[2 * BT + o] = act;
    }
    asm volatile("s_wait_dscnt 0x0" ::: "memory");  // ex reads done before next-step hs writes
  }
}

extern "C" void kernel_launch(void* const* d_in, const int* in_sizes, int n_in,
                              void* d_out, int out_size, void* d_ws, size_t ws_size,
                              hipStream_t stream) {
  const float* sampled_z   = (const float*)d_in[0];
  const float* idm_s       = (const float*)d_in[1];
  const float* merger_cs   = (const float*)d_in[2];
  const float* noise       = (const float*)d_in[3];
  const float* scaler_mean = (const float*)d_in[4];
  const float* scaler_std  = (const float*)d_in[5];
  const float* W1          = (const float*)d_in[6];
  const float* b1          = (const float*)d_in[7];
  const float* W2          = (const float*)d_in[8];
  const float* b2          = (const float*)d_in[9];
  const float* Wm          = (const float*)d_in[10];
  const float* bm          = (const float*)d_in[11];
  const float* Wv          = (const float*)d_in[12];
  const float* bv          = (const float*)d_in[13];
  float* out = (float*)d_out;

  dim3 grid(B_TOT / (16 * WAVES));   // 2048 wave-tiles / 8 waves per block = 256 blocks
  dim3 block(256);
  ForwardSim_kernel<<<grid, block, 0, stream>>>(
      sampled_z, idm_s, merger_cs, noise, scaler_mean, scaler_std,
      W1, b1, W2, b2, Wm, bm, Wv, bv, out);
}
